// BayesianSSMKernel_58746562674865
// MI455X (gfx1250) — matrix operand order repaired and checked by reading, not compiled, not run on hardware
//
#include <hip/hip_runtime.h>

// ---------------- problem constants ----------------
#define B_ 4
#define L_ 4096
#define D_ 1024
#define N_ 1024
#define M_ (B_ * L_)          // 16384 rows
#define CH_ 16                // scan chunks per channel
#define CL_ (L_ / CH_)        // 256 steps per chunk

// GEMM tiling: block = 4 waves, each wave computes a 32x64 strip
#define BM 128
#define BN 64
#define BK 32

typedef __attribute__((ext_vector_type(16))) __bf16 v16bf;
typedef __attribute__((ext_vector_type(8)))  float  v8f;

union Frag16 { v16bf v; unsigned int u[8]; };

// ---------------- helpers ----------------
static __device__ __forceinline__ unsigned int pcg_hash(unsigned int v) {
    unsigned int s = v * 747796405u + 2891336453u;
    unsigned int w = ((s >> ((s >> 28) + 4u)) ^ s) * 277803737u;
    return (w >> 22) ^ w;
}

static __device__ __forceinline__ float gauss_from(unsigned int s) {
    unsigned int a = pcg_hash(s);
    unsigned int b = pcg_hash(s ^ 0x68bc21ebu);
    float u1 = (float)(a >> 8) * (1.0f / 16777216.0f) + 1.0e-7f;
    float u2 = (float)(b >> 8) * (1.0f / 16777216.0f);
    return sqrtf(-2.0f * __logf(u1)) * __cosf(6.28318530718f * u2);
}

static __device__ __forceinline__ unsigned short f2bf(float f) {
    unsigned int u = __float_as_uint(f);
    unsigned int r = (u + 0x7FFFu + ((u >> 16) & 1u)) >> 16;
    return (unsigned short)r;
}

// LDS byte address = low 32 bits of the generic pointer (aperture rule:
// generic address bits 31:0 give the LDS address)
static __device__ __forceinline__ unsigned int lds_addr32(const void* p) {
    return (unsigned int)(size_t)p;
}

// Async DMA: global -> LDS, 16 bytes per lane, tracked on ASYNCcnt.
// saddr form: mem_addr = SGPR_base + VGPR_u32_offset
static __device__ __forceinline__ void async_ld_b128(unsigned int lds_off,
                                                     unsigned int gbyte_off,
                                                     const void* base) {
    asm volatile("global_load_async_to_lds_b128 %0, %1, %2"
                 :: "v"(lds_off), "v"(gbyte_off), "s"(base)
                 : "memory");
}

static __device__ __forceinline__ void wait_async0() {
    asm volatile("s_wait_asynccnt 0x0" ::: "memory");
}

// ---------------- 1. sample Bayesian weights (f32 -> bf16) ----------------
__global__ __launch_bounds__(256)
void sample_w_kernel(const float* __restrict__ mu, const float* __restrict__ lv,
                     unsigned short* __restrict__ out, unsigned int seed, int n) {
    int i = blockIdx.x * blockDim.x + threadIdx.x;
    if (i < n) {
        float w = mu[i] + __expf(0.5f * lv[i]) * gauss_from(seed + (unsigned int)i * 2654435761u);
        out[i] = f2bf(w);
    }
}

__global__ __launch_bounds__(256)
void sample_b_kernel(const float* __restrict__ mu, const float* __restrict__ lv,
                     float* __restrict__ out, unsigned int seed, int n) {
    int i = blockIdx.x * blockDim.x + threadIdx.x;
    if (i < n)
        out[i] = mu[i] + __expf(0.5f * lv[i]) * gauss_from(seed + (unsigned int)i * 2654435761u);
}

// ---------------- 2. x f32 -> bf16 ----------------
__global__ __launch_bounds__(256)
void convert_x_kernel(const float* __restrict__ x, unsigned short* __restrict__ xb) {
    int i = (blockIdx.x * blockDim.x + threadIdx.x) * 4;
    float4 v = *(const float4*)(x + i);
    ushort4 o;
    o.x = f2bf(v.x); o.y = f2bf(v.y); o.z = f2bf(v.z); o.w = f2bf(v.w);
    *(ushort4*)(xb + i) = o;
}

// ---------------- 3. WMMA GEMM: P[p] = x @ W[p]^T + b[p] ----------------
// grid: (N/BN, M/BM, 4 projections); block: 128 threads = 4 waves (wave32).
// Double-buffered LDS tiles filled by the async DMA engine (ASYNCcnt);
// K loop manually unrolled x2 so the buffer index is a literal constant.
__global__ __launch_bounds__(128)
void gemm_bf16_wmma_kernel(const unsigned short* __restrict__ xb,     // [M,D] bf16
                           const unsigned short* __restrict__ wball,  // [4][N,D] bf16
                           const float* __restrict__ biasall,         // [4][N]
                           float* __restrict__ Pall)                  // [4][M,N]
{
    const int p = blockIdx.z;
    const unsigned short* wb = wball + (size_t)p * (size_t)N_ * D_;
    const float* bias = biasall + p * N_;
    float* P = Pall + (size_t)p * (size_t)M_ * N_;

    __shared__ unsigned short xs[2][BM * BK];   // 2 x 8 KB
    __shared__ unsigned short wsm[2][BN * BK];  // 2 x 4 KB

    const int tid  = threadIdx.x;
    const int wave = tid >> 5;
    const int lane = tid & 31;
    const int r    = lane & 15;
    const bool hi  = lane >= 16;

    const int m0 = blockIdx.y * BM;
    const int n0 = blockIdx.x * BN;

    // staging: 16B chunks; xs tile = 512 chunks (4/thread), wsm = 256 (2/thread)
    unsigned int xoff[4], lxs[2][4];
#pragma unroll
    for (int i = 0; i < 4; ++i) {
        const int c = tid + i * 128;
        const int row = c >> 2, col = (c & 3) << 3;
        xoff[i]   = ((unsigned int)(m0 + row) * D_ + col) * 2u;
        lxs[0][i] = lds_addr32(&xs[0][row * BK + col]);
        lxs[1][i] = lds_addr32(&xs[1][row * BK + col]);
    }
    unsigned int woff[2], lws[2][2];
#pragma unroll
    for (int i = 0; i < 2; ++i) {
        const int c = tid + i * 128;
        const int row = c >> 2, col = (c & 3) << 3;
        woff[i]   = ((unsigned int)(n0 + row) * D_ + col) * 2u;
        lws[0][i] = lds_addr32(&wsm[0][row * BK + col]);
        lws[1][i] = lds_addr32(&wsm[1][row * BK + col]);
    }

    v8f acc[2][4] = {};

    // one pipeline stage; `buf` must be a compile-time literal at each call
    // site so that all LDS-address array indexing constant-folds.
    auto pipe_step = [&](const int buf, const int k0) {
        wait_async0();        // our DMA into `buf` has landed in LDS
        __syncthreads();      // everyone's DMA landed; everyone done reading buf^1

        if (k0 + BK < D_) {   // DMA next K tile into the other buffer
            const unsigned int kadd = (unsigned int)(k0 + BK) * 2u;
#pragma unroll
            for (int i = 0; i < 4; ++i) async_ld_b128(lxs[buf ^ 1][i], xoff[i] + kadd, xb);
#pragma unroll
            for (int i = 0; i < 2; ++i) async_ld_b128(lws[buf ^ 1][i], woff[i] + kadd, wb);
        }

        // A fragments (16x32 bf16): lanes 0-15 row M=r hold K=0..15,
        // lanes 16-31 hold K=16..31
        Frag16 afr[2];
        const int kb = hi ? 8 : 0;
#pragma unroll
        for (int rt = 0; rt < 2; ++rt) {
            const unsigned int* xrow =
                (const unsigned int*)&xs[buf][(wave * 32 + rt * 16 + r) * BK];
#pragma unroll
            for (int j = 0; j < 8; ++j) {
                afr[rt].u[j]     = xrow[kb + j];
            }
        }

        // B fragments (32x16 bf16): lane column N=r; lanes 0-15 hold K=0..15,
        // lanes 16-31 hold K=16..31 — load all four, then WMMA burst
        Frag16 bfr[4];
        const int kb2 = hi ? 8 : 0;
#pragma unroll
        for (int t = 0; t < 4; ++t) {
            const unsigned int* wrow = (const unsigned int*)&wsm[buf][(t * 16 + r) * BK];
#pragma unroll
            for (int j = 0; j < 8; ++j) bfr[t].u[j] = wrow[kb2 + j];
        }

#pragma unroll
        for (int t = 0; t < 4; ++t)
#pragma unroll
            for (int rt = 0; rt < 2; ++rt)
                acc[rt][t] = __builtin_amdgcn_wmma_f32_16x16x32_bf16(
                    /*neg_a=*/false, afr[rt].v, /*neg_b=*/false, bfr[t].v,
                    /*c_mod=*/(short)0, acc[rt][t],
                    /*reuse_a=*/false, /*reuse_b=*/false);
    };

    // prologue: async-fill buffer 0 with the k0 = 0 tiles
#pragma unroll
    for (int i = 0; i < 4; ++i) async_ld_b128(lxs[0][i], xoff[i], xb);
#pragma unroll
    for (int i = 0; i < 2; ++i) async_ld_b128(lws[0][i], woff[i], wb);

    // unroll x2: buffer index is a literal 0 / 1 in each stage
    for (int k0 = 0; k0 < D_; k0 += 2 * BK) {
        pipe_step(0, k0);
        pipe_step(1, k0 + BK);
    }

    // epilogue: C/D layout — VGPR rr: lanes 0-15 -> M=rr, lanes 16-31 -> M=8+rr
#pragma unroll
    for (int rt = 0; rt < 2; ++rt) {
        const int mb = m0 + wave * 32 + rt * 16 + (hi ? 8 : 0);
#pragma unroll
        for (int t = 0; t < 4; ++t) {
            const int col = n0 + t * 16 + r;
            const float bv = bias[col];
#pragma unroll
            for (int rr = 0; rr < 8; ++rr)
                P[(size_t)(mb + rr) * N_ + col] = acc[rt][t][rr] + bv;
        }
    }
}

// ---------------- 4. fused SSM epilogue: E and Bx in place ----------------
__global__ __launch_bounds__(256)
void ssm_epilogue_kernel(float* __restrict__ PA, const float* __restrict__ Pdt,
                         float* __restrict__ PB, const float* __restrict__ x,
                         size_t total) {
    size_t i = (size_t)blockIdx.x * blockDim.x + threadIdx.x;
    size_t stride = (size_t)gridDim.x * blockDim.x;
    for (; i < total; i += stride) {
        // E = exp(A*dt) = exp(-exp(linA)*exp(lindt)) = exp(-exp(linA+lindt))
        float s = PA[i] + Pdt[i];
        PA[i] = __expf(-__expf(s));
        PB[i] = PB[i] * x[i];      // Bx (N == D broadcast)
    }
}

// ---------------- 5. chunked parallel scan (3 passes) ----------------
// pass1: per (channel, chunk) compute composed affine (prodE, h_local_end)
__global__ __launch_bounds__(256)
void scan_pass1_kernel(const float* __restrict__ E, const float* __restrict__ Bx,
                       float* __restrict__ cA, float* __restrict__ cB) {
    int tid = blockIdx.x * blockDim.x + threadIdx.x;   // 0..65535
    int n = tid & (N_ - 1);
    int rest = tid >> 10;
    int b = rest & (B_ - 1);
    int j = rest >> 2;
    size_t base = ((size_t)(b * L_) + (size_t)j * CL_) * (size_t)N_ + n;
    float aP = 1.0f, h = 0.0f;
    for (int l = 0; l < CL_; ++l) {
        float e = E[base], bx = Bx[base];
        aP *= e;
        h = h * e + bx;
        base += N_;
    }
    size_t o = ((size_t)j * B_ + b) * N_ + n;
    cA[o] = aP; cB[o] = h;
}

// pass2: sequential combine of CH_ chunks per channel, store chunk-start states
__global__ __launch_bounds__(256)
void scan_pass2_kernel(const float* __restrict__ cA, const float* __restrict__ cB,
                       float* __restrict__ Hst) {
    int tid = blockIdx.x * blockDim.x + threadIdx.x;   // 0..4095
    int n = tid & (N_ - 1);
    int b = tid >> 10;
    float H = 0.0f;
    for (int j = 0; j < CH_; ++j) {
        size_t o = ((size_t)j * B_ + b) * N_ + n;
        Hst[o] = H;
        H = cA[o] * H + cB[o];
    }
}

// pass3: replay each chunk with its true h0 and emit y = C*h*batch
__global__ __launch_bounds__(256)
void scan_pass3_kernel(const float* __restrict__ E, const float* __restrict__ Bx,
                       const float* __restrict__ Craw, const float* __restrict__ Hst,
                       float* __restrict__ out) {
    int tid = blockIdx.x * blockDim.x + threadIdx.x;   // 0..65535
    int n = tid & (N_ - 1);
    int rest = tid >> 10;
    int b = rest & (B_ - 1);
    int j = rest >> 2;
    float h = Hst[((size_t)j * B_ + b) * N_ + n];
    size_t base = ((size_t)(b * L_) + (size_t)j * CL_) * (size_t)N_ + n;
    for (int l = 0; l < CL_; ++l) {
        // speculative prefetch a few steps ahead (global_prefetch_b8)
        __builtin_prefetch(&E[base + (size_t)8 * N_], 0, 0);
        __builtin_prefetch(&Craw[base + (size_t)8 * N_], 0, 0);
        float e = E[base], bx = Bx[base], c = Craw[base];
        h = h * e + bx;
        out[base] = c * h * (float)B_;
        base += N_;
    }
}

// ---------------- launcher ----------------
extern "C" void kernel_launch(void* const* d_in, const int* in_sizes, int n_in,
                              void* d_out, int out_size, void* d_ws, size_t ws_size,
                              hipStream_t stream) {
    (void)in_sizes; (void)n_in; (void)out_size; (void)ws_size;

    const float* x = (const float*)d_in[0];
    // input order: x, A(1-4), Bp(5-8), C(9-12), dt(13-16); each (wmu,wlv,bmu,blv)
    // projection order p: 0=A, 1=dt, 2=Bp, 3=C
    const float* wmu[4] = { (const float*)d_in[1],  (const float*)d_in[13],
                            (const float*)d_in[5],  (const float*)d_in[9] };
    const float* wlv[4] = { (const float*)d_in[2],  (const float*)d_in[14],
                            (const float*)d_in[6],  (const float*)d_in[10] };
    const float* bmu[4] = { (const float*)d_in[3],  (const float*)d_in[15],
                            (const float*)d_in[7],  (const float*)d_in[11] };
    const float* blv[4] = { (const float*)d_in[4],  (const float*)d_in[16],
                            (const float*)d_in[8],  (const float*)d_in[12] };

    char* ws = (char*)d_ws;
    size_t off = 0;
    unsigned short* xb = (unsigned short*)(ws + off); off += (size_t)M_ * D_ * 2;        // 32 MB
    unsigned short* wb = (unsigned short*)(ws + off); off += (size_t)4 * N_ * D_ * 2;    //  8 MB
    float* bias = (float*)(ws + off);                 off += (size_t)4 * N_ * 4;
    float* P    = (float*)(ws + off);                 off += (size_t)4 * M_ * N_ * 4;    // 256 MB
    float* cA   = (float*)(ws + off);                 off += (size_t)CH_ * B_ * N_ * 4;
    float* cB   = (float*)(ws + off);                 off += (size_t)CH_ * B_ * N_ * 4;
    float* Hst  = (float*)(ws + off);                 off += (size_t)CH_ * B_ * N_ * 4;

    float* out = (float*)d_out;

    const int NW = N_ * D_;   // 1,048,576 weights per layer
    for (int p = 0; p < 4; ++p) {
        sample_w_kernel<<<(NW + 255) / 256, 256, 0, stream>>>(
            wmu[p], wlv[p], wb + (size_t)p * NW, 0x1234567u + (unsigned)p * 0x9E3779B9u, NW);
        sample_b_kernel<<<(N_ + 255) / 256, 256, 0, stream>>>(
            bmu[p], blv[p], bias + p * N_, 0xABCDEF1u + (unsigned)p * 0x85EBCA6Bu, N_);
    }

    convert_x_kernel<<<(M_ * D_) / (256 * 4), 256, 0, stream>>>(x, xb);

    dim3 ggrid(N_ / BN, M_ / BM, 4);
    gemm_bf16_wmma_kernel<<<ggrid, 128, 0, stream>>>(xb, wb, bias, P);

    float* PA  = P;                               // raw linA -> E (in place)
    float* Pdt = P + (size_t)1 * M_ * N_;         // raw lindt
    float* PB  = P + (size_t)2 * M_ * N_;         // raw linB -> Bx (in place)
    float* PC  = P + (size_t)3 * M_ * N_;         // raw linC
    ssm_epilogue_kernel<<<4096, 256, 0, stream>>>(PA, Pdt, PB, x, (size_t)M_ * N_);

    scan_pass1_kernel<<<(CH_ * B_ * N_) / 256, 256, 0, stream>>>(PA, PB, cA, cB);
    scan_pass2_kernel<<<(B_ * N_) / 256, 256, 0, stream>>>(cA, cB, Hst);
    scan_pass3_kernel<<<(CH_ * B_ * N_) / 256, 256, 0, stream>>>(PA, PB, PC, Hst, out);
}
